// gat_49417893707850
// MI455X (gfx1250) — matrix operand order repaired and checked by reading, not compiled, not run on hardware
//
#include <hip/hip_runtime.h>
#include <hip/hip_bf16.h>
#include <stdint.h>

// ---------------------------------------------------------------------------
// CDNA5 (gfx1250) wave32 WMMA types
// ---------------------------------------------------------------------------
typedef float v8f  __attribute__((ext_vector_type(8)));
typedef __bf16 v16bf __attribute__((ext_vector_type(16)));
typedef unsigned int u32x4 __attribute__((ext_vector_type(4)));
typedef int i32x8 __attribute__((ext_vector_type(8)));
typedef int i32x4 __attribute__((ext_vector_type(4)));

union Frag {
    uint32_t u[8];
    uint4    q[2];
    v16bf    v;
};

__device__ __forceinline__ uint16_t f2bf(float f) {
    uint32_t u = __float_as_uint(f);
    u += 0x7FFFu + ((u >> 16) & 1u);          // round-to-nearest-even
    return (uint16_t)(u >> 16);
}
__device__ __forceinline__ float bf2f(uint16_t h) {
    return __uint_as_float(((uint32_t)h) << 16);
}

#define NNODE 8192

#if __has_builtin(__builtin_amdgcn_tensor_load_to_lds) && \
    __has_builtin(__builtin_amdgcn_s_wait_tensorcnt)
#define HAVE_TDM 1
#else
#define HAVE_TDM 0
#endif

// ---------------------------------------------------------------------------
// Stage the 32KB f2 vector into LDS.
// TDM path: one tensor_load_to_lds DMA issued by wave 0 (1-D tile, 8192 x f32,
// D# built per ISA 8.3/8.4 bitfields), wait on TENSORcnt, then block barrier.
// This toolchain exposes the 6-arg builtin (clang-23 / therock-10.0 form):
//   (u32x4 group0, i32x8 group1, i32x4 group2, i32x4 group3, i32x8, i32 cpol)
// Fallback: cooperative vector loads.
// NOTE: s_f2 must be the first __shared__ object (descriptor lds_addr = 0).
// ---------------------------------------------------------------------------
__device__ __forceinline__ void stage_f2(const float* __restrict__ f2,
                                         float* s_f2)
{
#if HAVE_TDM
    if (threadIdx.x < 32) {
        uint64_t ga = (uint64_t)(uintptr_t)f2;
        u32x4 g0;
        g0[0] = 1u;                              // count=1, user descriptor
        g0[1] = 0u;                              // lds_addr = 0
        g0[2] = (uint32_t)ga;                    // global_addr[31:0]
        g0[3] = (uint32_t)((ga >> 32) & 0x01FFFFFFu) | 0x80000000u; // [56:32] | type=2
        i32x8 g1;
        g1[0] = 0x00020000;                      // data_size = 4B (bits 17:16)
        g1[1] = (int)(8192u << 16);              // tensor_dim0 = 8192 (bits 79:48)
        g1[2] = (int)(1u << 16);                 // tensor_dim1 = 1   (bits 111:80)
        g1[3] = (int)(8192u << 16);              // tile_dim0 = 8192  (bits 127:112)
        g1[4] = 1;                               // tile_dim1 = 1
        g1[5] = 8192;                            // tensor_dim0_stride
        g1[6] = 0;
        g1[7] = 0;
        i32x4 z4 = {0, 0, 0, 0};
        i32x8 z8 = {0, 0, 0, 0, 0, 0, 0, 0};
        __builtin_amdgcn_tensor_load_to_lds(g0, g1, z4, z4, z8, 0);
        __builtin_amdgcn_s_wait_tensorcnt(0);
    }
    __syncthreads();
#else
    for (int k = threadIdx.x; k < NNODE; k += 256) s_f2[k] = f2[k];
    __syncthreads();
#endif
}

// ---------------------------------------------------------------------------
// 1) Convert x -> bf16, W_heads -> WbT (transposed, head-concat), W_out -> WoT
//    x is streamed once: non-temporal loads keep it out of L2.
// ---------------------------------------------------------------------------
__global__ __launch_bounds__(256) void k_convert(
    const float* __restrict__ x, const float* __restrict__ Whd,
    const float* __restrict__ Wod,
    uint16_t* __restrict__ xb, uint16_t* __restrict__ WbT,
    uint16_t* __restrict__ WoT)
{
    int idx = blockIdx.x * 256 + threadIdx.x;
    if (idx < NNODE * 512) xb[idx] = f2bf(__builtin_nontemporal_load(x + idx));
    if (idx < 256 * 512) {                       // WbT[o][k], o = head*128+c
        int o = idx >> 9, k = idx & 511;
        int h = o >> 7, c = o & 127;
        WbT[idx] = f2bf(Whd[h * 512 * 128 + k * 128 + c]);
    }
    if (idx < 64 * 256) {                        // WoT[o][k] = W_out[k][o]
        int o = idx >> 8, k = idx & 255;
        WoT[idx] = f2bf(Wod[k * 64 + o]);
    }
}

// ---------------------------------------------------------------------------
// 2) adj (int32 NxN, 256MB read-once) -> bitmask (8MB, L2-resident).
//    wave32 ballot; non-temporal loads so adj never pollutes L2.
// ---------------------------------------------------------------------------
__global__ __launch_bounds__(256) void k_pack_adj(
    const int* __restrict__ adj, uint32_t* __restrict__ bits)
{
    size_t t = (size_t)blockIdx.x * 256 + threadIdx.x;
    int v = __builtin_nontemporal_load(adj + t);
    uint32_t m = __builtin_amdgcn_ballot_w32(v != 0);
    if ((threadIdx.x & 31) == 0) bits[t >> 5] = m;
}

// ---------------------------------------------------------------------------
// 3) Generic WMMA GEMM: OT[n][m] = sum_k A[m][k] * WT[n][k]  (bf16, f32 acc)
//    A row-major [M,K]; WT row-major [N,K]; output stored TRANSPOSED [N,M]
//    so it can serve as a B-operand (K-contiguous per column) downstream.
//    One 16x16 tile per wave, fragments loaded straight from L2 (no LDS).
// ---------------------------------------------------------------------------
__global__ __launch_bounds__(256) void k_gemm_at(
    const uint16_t* __restrict__ A, const uint16_t* __restrict__ WT,
    uint16_t* __restrict__ OT, int M, int N, int K)
{
    int wave = threadIdx.x >> 5, lane = threadIdx.x & 31;
    int tilesN = N >> 4;
    int gid = blockIdx.x * 8 + wave;
    int im  = (gid / tilesN) << 4;
    int in_ = (gid % tilesN) << 4;
    if (im >= M) return;
    int l15 = lane & 15, hb = lane >> 4;

    v8f acc = {};
    const uint16_t* arow = A  + (size_t)(im  + l15) * K;
    const uint16_t* brow = WT + (size_t)(in_ + l15) * K;
    for (int k0 = 0; k0 < K; k0 += 32) {
        Frag fa, fb;
        // A frag: lanes0-15 K=0..7 & 16..23, lanes16-31 +8
        fa.q[0] = *(const uint4*)(arow + k0 + hb * 8);
        fa.q[1] = *(const uint4*)(arow + k0 + hb * 8 + 16);
        // B frag: lanes0-15 K=0..7 & 16..23, lanes16-31 +8
        fb.q[0] = *(const uint4*)(brow + k0 + hb * 8);
        fb.q[1] = *(const uint4*)(brow + k0 + hb * 8 + 16);
        acc = __builtin_amdgcn_wmma_f32_16x16x32_bf16(
            false, fa.v, false, fb.v, (short)0, acc, false, false);
    }
    // store transposed: lane n writes m = hb*8 .. hb*8+7 packed as 8 bf16
    uint32_t pk[4];
#pragma unroll
    for (int p = 0; p < 4; ++p)
        pk[p] = (uint32_t)f2bf(acc[2 * p]) | ((uint32_t)f2bf(acc[2 * p + 1]) << 16);
    *(uint4*)(OT + (size_t)(in_ + l15) * M + im + hb * 8) = *(uint4*)pk;
}

// ---------------------------------------------------------------------------
// 4) f1/f2 vectors: f1[i] = WhT[:,i].a[0:F], f2[i] = WhT[:,i].a[F:2F]
// ---------------------------------------------------------------------------
__global__ __launch_bounds__(256) void k_compute_f(
    const uint16_t* __restrict__ WhT, const float* __restrict__ a,
    float* __restrict__ f1, float* __restrict__ f2, int F)
{
    int i = blockIdx.x * 256 + threadIdx.x;
    float s1 = 0.f, s2 = 0.f;
    for (int o = 0; o < F; ++o) {
        float v = bf2f(WhT[(size_t)o * NNODE + i]);
        s1 += a[o] * v;
        s2 += a[F + o] * v;
    }
    f1[i] = s1; f2[i] = s2;
}

// ---------------------------------------------------------------------------
// 5) Softmax stats per row: m_i = max over masked lrelu(f1_i+f2_j),
//    l_i = sum exp(s - m).  One wave per row; f2 TDM-staged into LDS.
// ---------------------------------------------------------------------------
__global__ __launch_bounds__(256) void k_att_stats(
    const uint32_t* __restrict__ bits, const float* __restrict__ f1,
    const float* __restrict__ f2, float* __restrict__ mrow,
    float* __restrict__ lrow)
{
    __shared__ float s_f2[NNODE];        // must stay the first LDS object
    stage_f2(f2, s_f2);

    int wave = threadIdx.x >> 5, lane = threadIdx.x & 31;
    int row = blockIdx.x * 8 + wave;
    float f1r = f1[row];
    const uint32_t* br = bits + (size_t)row * 256;

    float m = -INFINITY;
    for (int wi = lane; wi < 256; wi += 32) {
        uint32_t w = br[wi];
#pragma unroll 8
        for (int b = 0; b < 32; ++b) {
            float s = f1r + s_f2[wi * 32 + b];
            s = fmaxf(s, 0.2f * s);                       // leaky relu
            m = fmaxf(m, ((w >> b) & 1u) ? s : -INFINITY);
        }
    }
    for (int o = 16; o >= 1; o >>= 1) m = fmaxf(m, __shfl_xor(m, o, 32));

    float l = 0.f;
    if (m > -INFINITY) {
        for (int wi = lane; wi < 256; wi += 32) {
            uint32_t w = br[wi];
#pragma unroll 8
            for (int b = 0; b < 32; ++b) {
                float s = f1r + s_f2[wi * 32 + b];
                s = fmaxf(s, 0.2f * s);
                float se = ((w >> b) & 1u) ? s : -INFINITY;
                l += __expf(se - m);                      // exp(-inf)=0
            }
        }
    }
    for (int o = 16; o >= 1; o >>= 1) l += __shfl_xor(l, o, 32);
    if (lane == 0) { mrow[row] = m; lrow[row] = l; }
}

// ---------------------------------------------------------------------------
// 6) Fused attention GEMM: out = elu( softmax(mask, lrelu(f1_i+f2_j)) @ Wh )
//    Block = 16 rows x (NT*16) cols, 8 waves each own a j-stripe.
//    Each lane synthesizes its 16x32 bf16 A-fragment of probabilities
//    directly in WMMA A layout (16 exps/lane), B fragments are contiguous
//    b128 loads from the transposed WhT (L2-resident).
//    Wave partials reduced via ds_add_f32, elu fused into the store.
// ---------------------------------------------------------------------------
template <int NT>
__global__ __launch_bounds__(256) void k_att_gemm(
    const uint32_t* __restrict__ bits, const float* __restrict__ f1,
    const float* __restrict__ f2, const float* __restrict__ mrow,
    const float* __restrict__ lrow, const uint16_t* __restrict__ WhT,
    uint16_t* __restrict__ outb, int ldo, float* __restrict__ outf, int ldf)
{
    __shared__ float s_f2[NNODE];        // must stay the first LDS object
    __shared__ float s_acc[16 * NT * 16];
    for (int k = threadIdx.x; k < 16 * NT * 16; k += 256) s_acc[k] = 0.f;
    stage_f2(f2, s_f2);                  // contains the block barrier

    int wave = threadIdx.x >> 5, lane = threadIdx.x & 31;
    int l15 = lane & 15, hb = lane >> 4;
    int i0 = blockIdx.x * 16;
    int row = i0 + l15;
    float f1r = f1[row], mr = mrow[row], lr = lrow[row];
    float invl = (lr > 0.f) ? 1.f / lr : 0.f;
    const uint32_t* brr = bits + (size_t)row * 256;

    v8f acc[NT] = {};
    for (int t = 0; t < 32; ++t) {
        int jw = wave + 8 * t;          // this wave's 32-wide j block
        int j0 = jw << 5;
        uint32_t w = brr[jw];
        Frag fa;
#pragma unroll
        for (int p = 0; p < 8; ++p) {
            // A frag in contiguous k order (k = 16*hb + s, as B): dword p holds K = kk,kk+1
            int kk = 2 * p + hb * 16;
            float2 fp = *(const float2*)(s_f2 + j0 + kk);   // ds_read_b64
            float s0 = f1r + fp.x; s0 = fmaxf(s0, 0.2f * s0);
            float s1 = f1r + fp.y; s1 = fmaxf(s1, 0.2f * s1);
            float e0 = ((w >> kk) & 1u)       ? __expf(s0 - mr) * invl : 0.f;
            float e1 = ((w >> (kk + 1)) & 1u) ? __expf(s1 - mr) * invl : 0.f;
            fa.u[p] = (uint32_t)f2bf(e0) | ((uint32_t)f2bf(e1) << 16);
        }
#pragma unroll
        for (int c = 0; c < NT; ++c) {
            Frag fb;
            const uint16_t* bp =
                WhT + (size_t)(c * 16 + l15) * NNODE + j0 + hb * 16;
            fb.q[0] = *(const uint4*)bp;
            fb.q[1] = *(const uint4*)(bp + 8);
            acc[c] = __builtin_amdgcn_wmma_f32_16x16x32_bf16(
                false, fa.v, false, fb.v, (short)0, acc[c], false, false);
        }
    }
    // reduce the 8 wave partials through LDS (ds_add_f32)
#pragma unroll
    for (int c = 0; c < NT; ++c)
#pragma unroll
        for (int r = 0; r < 8; ++r)
            atomicAdd(&s_acc[(r + hb * 8) * (NT * 16) + c * 16 + l15],
                      acc[c][r]);
    __syncthreads();

    const int Fout = NT * 16;
    for (int e = threadIdx.x; e < 16 * Fout; e += 256) {
        int mm = e / Fout, col = e % Fout;
        float v = s_acc[e];
        v = (v > 0.f) ? v : (__expf(v) - 1.f);            // elu
        outb[(size_t)(i0 + mm) * ldo + col] = f2bf(v);
        if (outf) outf[(size_t)(i0 + mm) * ldf + col] = v;
    }
}

// ---------------------------------------------------------------------------
// 7) reconstruction = mu @ mu.T  (B^T == mu, so both operands are row-major
//    mu tiles; K=64 -> 2 WMMAs per 16x16 tile).  256MB write-once output:
//    non-temporal stores keep it from evicting the L2-resident working set.
// ---------------------------------------------------------------------------
__global__ __launch_bounds__(256) void k_recon(
    const uint16_t* __restrict__ mub, float* __restrict__ out)
{
    int wave = threadIdx.x >> 5, lane = threadIdx.x & 31;
    int gid = blockIdx.x * 8 + wave;
    int i0 = (gid >> 9) << 4;
    int j0 = (gid & 511) << 4;
    int l15 = lane & 15, hb = lane >> 4;

    v8f acc = {};
    const uint16_t* ar = mub + (size_t)(i0 + l15) * 64;
    const uint16_t* br = mub + (size_t)(j0 + l15) * 64;
#pragma unroll
    for (int k0 = 0; k0 < 64; k0 += 32) {
        Frag fa, fb;
        fa.q[0] = *(const uint4*)(ar + k0 + hb * 8);
        fa.q[1] = *(const uint4*)(ar + k0 + hb * 8 + 16);
        fb.q[0] = *(const uint4*)(br + k0 + hb * 8);
        fb.q[1] = *(const uint4*)(br + k0 + hb * 8 + 16);
        acc = __builtin_amdgcn_wmma_f32_16x16x32_bf16(
            false, fa.v, false, fb.v, (short)0, acc, false, false);
    }
#pragma unroll
    for (int r = 0; r < 8; ++r)
        __builtin_nontemporal_store(
            acc[r], out + (size_t)(i0 + r + hb * 8) * NNODE + j0 + l15);
}

// ---------------------------------------------------------------------------
// 8) class_pre = softmax(mu @ W1) ; tiny, one thread per row (fp32)
// ---------------------------------------------------------------------------
__global__ __launch_bounds__(256) void k_classpre(
    const float* __restrict__ mu, const float* __restrict__ W1,
    float* __restrict__ out)
{
    int i = blockIdx.x * 256 + threadIdx.x;
    const float* mr = mu + (size_t)i * 64;
    float lg[16];
#pragma unroll
    for (int c = 0; c < 16; ++c) lg[c] = 0.f;
    for (int k = 0; k < 64; ++k) {
        float v = mr[k];
#pragma unroll
        for (int c = 0; c < 16; ++c) lg[c] += v * W1[k * 16 + c];
    }
    float mx = lg[0];
#pragma unroll
    for (int c = 1; c < 16; ++c) mx = fmaxf(mx, lg[c]);
    float s = 0.f;
#pragma unroll
    for (int c = 0; c < 16; ++c) { lg[c] = __expf(lg[c] - mx); s += lg[c]; }
    float inv = 1.f / s;
#pragma unroll
    for (int c = 0; c < 16; ++c) out[(size_t)i * 16 + c] = lg[c] * inv;
}

// ---------------------------------------------------------------------------
// host driver
// ---------------------------------------------------------------------------
extern "C" void kernel_launch(void* const* d_in, const int* in_sizes, int n_in,
                              void* d_out, int out_size, void* d_ws,
                              size_t ws_size, hipStream_t stream)
{
    (void)in_sizes; (void)n_in; (void)out_size; (void)ws_size;
    const float* x    = (const float*)d_in[0];
    const int*   adj  = (const int*)d_in[1];
    const float* Whd  = (const float*)d_in[2];   // [2,512,128]
    const float* ahd  = (const float*)d_in[3];   // [2,256,1]
    const float* Wod  = (const float*)d_in[4];   // [256,64]
    const float* aod  = (const float*)d_in[5];   // [128,1]
    const float* W1   = (const float*)d_in[6];   // [64,16]

    const int N = NNODE;
    float* out   = (float*)d_out;
    float* recon = out;                               // [N,N]
    float* muF   = out + (size_t)N * N;               // [N,64]
    float* cls   = muF + (size_t)N * 64;              // [N,16]

    char* ws = (char*)d_ws;
    size_t off = 0;
    auto alloc = [&](size_t bytes) -> void* {
        void* p = ws + off;
        off = (off + bytes + 255) & ~(size_t)255;
        return p;
    };
    uint16_t* xb   = (uint16_t*)alloc((size_t)N * 512 * 2);   // x bf16
    uint32_t* bits = (uint32_t*)alloc((size_t)N * 256 * 4);   // adj bitmask
    uint16_t* WbT  = (uint16_t*)alloc((size_t)256 * 512 * 2); // W heads ^T
    uint16_t* WoT  = (uint16_t*)alloc((size_t)64 * 256 * 2);  // W_out ^T
    uint16_t* WhT  = (uint16_t*)alloc((size_t)256 * N * 2);   // (x@W)^T
    uint16_t* hb   = (uint16_t*)alloc((size_t)N * 256 * 2);   // layer1 out
    uint16_t* Wh2T = (uint16_t*)alloc((size_t)64 * N * 2);    // (h@Wo)^T
    uint16_t* mub  = (uint16_t*)alloc((size_t)N * 64 * 2);    // mu bf16
    float* f1h0 = (float*)alloc((size_t)N * 4);
    float* f2h0 = (float*)alloc((size_t)N * 4);
    float* f1h1 = (float*)alloc((size_t)N * 4);
    float* f2h1 = (float*)alloc((size_t)N * 4);
    float* m0   = (float*)alloc((size_t)N * 4);
    float* l0   = (float*)alloc((size_t)N * 4);
    float* m1   = (float*)alloc((size_t)N * 4);
    float* l1   = (float*)alloc((size_t)N * 4);
    float* f1o  = (float*)alloc((size_t)N * 4);
    float* f2o  = (float*)alloc((size_t)N * 4);
    float* mo   = (float*)alloc((size_t)N * 4);
    float* lo   = (float*)alloc((size_t)N * 4);

    // 1) conversions
    k_convert<<<(N * 512) / 256, 256, 0, stream>>>(x, Whd, Wod, xb, WbT, WoT);
    // 2) adj -> bitmask (single streaming pass over the 256MB matrix)
    k_pack_adj<<<(int)(((size_t)N * N) / 256), 256, 0, stream>>>(adj, bits);
    // 3) Wh = x @ W (both heads), stored transposed [256][N]
    k_gemm_at<<<(512 * 16) / 8, 256, 0, stream>>>(xb, WbT, WhT, N, 256, 512);
    // 4) attention score vectors per head
    k_compute_f<<<N / 256, 256, 0, stream>>>(WhT, ahd, f1h0, f2h0, 128);
    k_compute_f<<<N / 256, 256, 0, stream>>>(WhT + (size_t)128 * N, ahd + 256,
                                             f1h1, f2h1, 128);
    // 5) softmax stats
    k_att_stats<<<N / 8, 256, 0, stream>>>(bits, f1h0, f2h0, m0, l0);
    k_att_stats<<<N / 8, 256, 0, stream>>>(bits, f1h1, f2h1, m1, l1);
    // 6) h = concat_head elu(att @ Wh)
    k_att_gemm<8><<<N / 16, 256, 0, stream>>>(bits, f1h0, f2h0, m0, l0,
                                              WhT, hb, 256, (float*)nullptr, 0);
    k_att_gemm<8><<<N / 16, 256, 0, stream>>>(bits, f1h1, f2h1, m1, l1,
                                              WhT + (size_t)128 * N,
                                              hb + 128, 256, (float*)nullptr, 0);
    // 7) Wh2 = h @ W_out, stored transposed [64][N]
    k_gemm_at<<<(512 * 4) / 8, 256, 0, stream>>>(hb, WoT, Wh2T, N, 64, 256);
    // 8) layer-2 attention
    k_compute_f<<<N / 256, 256, 0, stream>>>(Wh2T, aod, f1o, f2o, 64);
    k_att_stats<<<N / 8, 256, 0, stream>>>(bits, f1o, f2o, mo, lo);
    // 9) mu = elu(att2 @ Wh2): bf16 for the recon GEMM + fp32 to d_out
    k_att_gemm<4><<<N / 16, 256, 0, stream>>>(bits, f1o, f2o, mo, lo,
                                              Wh2T, mub, 64, muF, 64);
    // 10) reconstruction = mu @ mu.T (store-bound: 256MB, NT stores)
    k_recon<<<(512 * 512) / 8, 256, 0, stream>>>(mub, recon);
    // 11) class_pre = softmax(mu @ W1)
    k_classpre<<<N / 256, 256, 0, stream>>>(muF, W1, cls);
}
